// SPDGTransformer_48593259987329
// MI455X (gfx1250) — matrix operand order repaired and checked by reading, not compiled, not run on hardware
//
#include <hip/hip_runtime.h>
#include <cstdint>
#include <cstddef>

// ---------------- problem constants (match reference) ----------------
#define VOCAB_SZ 32000
#define D_MODEL  512
#define N_HEADS  8
#define N_LAYERS 6
#define SEQ_LEN  1024
#define BATCH    4
#define D_HEAD   64            // D_MODEL / N_HEADS
#define WIN      102           // band half-width
#define M_ROWS   (BATCH * SEQ_LEN)   // 4096
#define NEG_INF  (-1e30f)

typedef _Float16 f16_t;
typedef __attribute__((ext_vector_type(16))) _Float16 v16h;
typedef __attribute__((ext_vector_type(8)))  float    v8f;

// ---------------- embedding: x = tok_emb[ids] + pos_emb ----------------
__global__ void embed_kernel(const int* __restrict__ ids,
                             const float* __restrict__ tok_emb,
                             const float* __restrict__ pos_emb,
                             float* __restrict__ x)
{
    int t = blockIdx.x * blockDim.x + threadIdx.x;
    if (t >= M_ROWS * D_MODEL) return;
    int d  = t % D_MODEL;
    int bs = t / D_MODEL;
    int s  = bs % SEQ_LEN;
    int tok = ids[bs];
    x[t] = tok_emb[(size_t)tok * D_MODEL + d] + pos_emb[(size_t)s * D_MODEL + d];
}

// ---------------- fp32 -> f16 conversion ----------------
__global__ void cvt_f16_kernel(const float* __restrict__ in,
                               f16_t* __restrict__ out, int n)
{
    int t = blockIdx.x * blockDim.x + threadIdx.x;
    if (t < n) out[t] = (f16_t)in[t];
}

// ---------------- weight transpose+convert: Wt[n,k] = (f16) W[k,n] ----------------
__global__ void wtrans_kernel(const float* __restrict__ W,  // [K=D, N=D] row-major
                              f16_t* __restrict__ Wt)       // [N, K] row-major
{
    int t = blockIdx.x * blockDim.x + threadIdx.x;
    if (t >= D_MODEL * D_MODEL) return;
    int k = t % D_MODEL;
    int n = t / D_MODEL;
    Wt[(size_t)n * D_MODEL + k] = (f16_t)W[(size_t)k * D_MODEL + n];
}

// ---------------- WMMA GEMM: C[M,N] = A[M,K] * Bt[N,K]^T + bias ----------------
// One wave computes a 32x32 macro-tile (2x2 WMMA tiles): 4 independent
// accumulator chains per wave (hides WMMA->WMMA RAW latency) and each
// fragment load feeds two v_wmma_f32_16x16x32_f16 ops (2x data reuse).
// Fragment layouts per CDNA5 ISA §7.12.2.
__global__ __launch_bounds__(256)
void gemm_wmma_kernel(const f16_t* __restrict__ A,   // [M,K]
                      const f16_t* __restrict__ Bt,  // [N,K]  (B transposed)
                      const float* __restrict__ bias,// [N] or nullptr
                      float* __restrict__ C,         // [M,N]
                      int M, int N, int K)
{
    int wave = (blockIdx.x * blockDim.x + threadIdx.x) >> 5;
    int lane = threadIdx.x & 31;
    int ntn  = N >> 5;                   // 32-col macro-tiles along N
    int tm   = wave / ntn;               // 32-row macro-tile index
    int tn   = wave % ntn;
    if (tm * 32 >= M) return;            // wave-uniform guard

    int lrow  = lane & 15;               // row (A) / col (B) within a 16-tile
    int khalf = lane >> 4;               // 0/1: K interleave half

    const f16_t* arow0 = A  + (size_t)(tm * 32 + lrow) * K;
    const f16_t* arow1 = arow0 + (size_t)16 * K;
    const f16_t* brow0 = Bt + (size_t)(tn * 32 + lrow) * K;
    const f16_t* brow1 = brow0 + (size_t)16 * K;

    union Frag { uint4 q[2]; v16h v; };

    v8f acc00 = {}, acc01 = {}, acc10 = {}, acc11 = {};
    for (int kb = 0; kb < K; kb += 32) {
        if (kb + 64 <= K) {              // prefetch next K block (global_prefetch_b8)
            __builtin_prefetch(arow0 + kb + 64, 0, 3);
            __builtin_prefetch(brow0 + kb + 64, 0, 3);
        }
        Frag a0, a1, b0, b1;
        // A 16x32 f16: lanes 0-15 hold K = kb+[0..7] and kb+[16..23];
        //              lanes 16-31 hold K = kb+[8..15] and kb+[24..31].
        a0.q[0] = *reinterpret_cast<const uint4*>(arow0 + kb + khalf * 8);
        a0.q[1] = *reinterpret_cast<const uint4*>(arow0 + kb + 16 + khalf * 8);
        a1.q[0] = *reinterpret_cast<const uint4*>(arow1 + kb + khalf * 8);
        a1.q[1] = *reinterpret_cast<const uint4*>(arow1 + kb + 16 + khalf * 8);
        // B 32x16 f16: same K interleave as A for its column.
        b0.q[0] = *reinterpret_cast<const uint4*>(brow0 + kb + khalf * 8);
        b0.q[1] = *reinterpret_cast<const uint4*>(brow0 + kb + 16 + khalf * 8);
        b1.q[0] = *reinterpret_cast<const uint4*>(brow1 + kb + khalf * 8);
        b1.q[1] = *reinterpret_cast<const uint4*>(brow1 + kb + 16 + khalf * 8);

        acc00 = __builtin_amdgcn_wmma_f32_16x16x32_f16(
                    false, a0.v, false, b0.v, (short)0, acc00, false, false);
        acc01 = __builtin_amdgcn_wmma_f32_16x16x32_f16(
                    false, a0.v, false, b1.v, (short)0, acc01, false, false);
        acc10 = __builtin_amdgcn_wmma_f32_16x16x32_f16(
                    false, a1.v, false, b0.v, (short)0, acc10, false, false);
        acc11 = __builtin_amdgcn_wmma_f32_16x16x32_f16(
                    false, a1.v, false, b1.v, (short)0, acc11, false, false);
    }

    // C/D layout: VGPR r -> row base + r + 8*khalf, col base + (lane&15)
    int c0 = tn * 32 + lrow;
    int r0 = tm * 32 + khalf * 8;
    float bv0 = bias ? bias[c0]      : 0.0f;
    float bv1 = bias ? bias[c0 + 16] : 0.0f;
    #pragma unroll
    for (int r = 0; r < 8; ++r) {
        C[(size_t)(r0 + r)      * N + c0]      = acc00[r] + bv0;
        C[(size_t)(r0 + r)      * N + c0 + 16] = acc01[r] + bv1;
        C[(size_t)(r0 + 16 + r) * N + c0]      = acc10[r] + bv0;
        C[(size_t)(r0 + 16 + r) * N + c0 + 16] = acc11[r] + bv1;
    }
}

// ---------------- per-token per-head gate: u = sigmoid(x @ Wu + bu) ----------------
__global__ void gate_kernel(const float* __restrict__ x,   // [M,D]
                            const float* __restrict__ Wu,  // [D,H]
                            const float* __restrict__ bu,  // [H]
                            float* __restrict__ u)         // [M,H]
{
    int t = blockIdx.x * blockDim.x + threadIdx.x;
    if (t >= M_ROWS * N_HEADS) return;
    int h = t % N_HEADS;
    int m = t / N_HEADS;
    float acc = bu[h];
    const float* xr = x + (size_t)m * D_MODEL;
    #pragma unroll 4
    for (int d = 0; d < D_MODEL; ++d)
        acc = fmaf(xr[d], Wu[(size_t)d * N_HEADS + h], acc);
    u[t] = 1.0f / (1.0f + __expf(-acc));
}

// ---------------- banded attention, online softmax ----------------
// One wave per (b, h, i); each lane owns 2 of the 64 head dims.
__global__ __launch_bounds__(256)
void attn_kernel(const float* __restrict__ q,   // [B*S, D] head-major
                 const float* __restrict__ k,
                 const float* __restrict__ v,
                 const float* __restrict__ u,   // [B*S, H]
                 float* __restrict__ out)       // [B*S, D]
{
    int wave = (blockIdx.x * blockDim.x + threadIdx.x) >> 5;
    int lane = threadIdx.x & 31;
    if (wave >= BATCH * N_HEADS * SEQ_LEN) return;
    int i = wave & (SEQ_LEN - 1);
    int h = (wave >> 10) & (N_HEADS - 1);
    int b = wave >> 13;

    const float scale = 0.125f;   // 1/sqrt(64)
    int d0 = lane * 2;

    const size_t rowq = (size_t)(b * SEQ_LEN + i) * D_MODEL + h * D_HEAD;
    float q0 = q[rowq + d0];
    float q1 = q[rowq + d0 + 1];

    int jlo = i - WIN; if (jlo < 0) jlo = 0;
    int jhi = i + WIN; if (jhi > SEQ_LEN - 1) jhi = SEQ_LEN - 1;

    float mrun = NEG_INF, lrun = 0.0f, o0 = 0.0f, o1 = 0.0f;
    for (int j = jlo; j <= jhi; ++j) {
        const size_t rowj = (size_t)(b * SEQ_LEN + j) * D_MODEL + h * D_HEAD;
        float part = q0 * k[rowj + d0] + q1 * k[rowj + d0 + 1];
        #pragma unroll
        for (int off = 16; off > 0; off >>= 1)
            part += __shfl_xor(part, off, 32);
        float s  = part * scale;
        float mn = fmaxf(mrun, s);
        float sc = __expf(mrun - mn);   // 0 on first iteration
        float p  = __expf(s - mn);
        lrun = lrun * sc + p;
        o0   = o0 * sc + p * v[rowj + d0];
        o1   = o1 * sc + p * v[rowj + d0 + 1];
        mrun = mn;
    }
    float inv = 1.0f / lrun;
    float g   = u[(size_t)(b * SEQ_LEN + i) * N_HEADS + h];
    out[rowq + d0]     = o0 * inv * g;
    out[rowq + d0 + 1] = o1 * inv * g;
}

// ---------------- residual + layernorm (one wave per 512-wide row) ----------------
__global__ __launch_bounds__(256)
void ln_kernel(const float* __restrict__ xin,
               const float* __restrict__ addin,   // nullptr => plain LN
               const float* __restrict__ g,
               const float* __restrict__ bta,
               float* __restrict__ out)
{
    int wave = (blockIdx.x * blockDim.x + threadIdx.x) >> 5;
    int lane = threadIdx.x & 31;
    if (wave >= M_ROWS) return;
    const size_t base = (size_t)wave * D_MODEL;

    float vals[16];
    float s = 0.0f;
    #pragma unroll
    for (int t = 0; t < 16; ++t) {
        int d = lane + 32 * t;
        float x = xin[base + d];
        if (addin) x += addin[base + d];
        vals[t] = x;
        s += x;
    }
    #pragma unroll
    for (int off = 16; off > 0; off >>= 1) s += __shfl_xor(s, off, 32);
    float mu = s * (1.0f / D_MODEL);

    float s2 = 0.0f;
    #pragma unroll
    for (int t = 0; t < 16; ++t) { float d = vals[t] - mu; s2 += d * d; }
    #pragma unroll
    for (int off = 16; off > 0; off >>= 1) s2 += __shfl_xor(s2, off, 32);
    float rstd = rsqrtf(s2 * (1.0f / D_MODEL) + 1e-5f);

    #pragma unroll
    for (int t = 0; t < 16; ++t) {
        int d = lane + 32 * t;
        out[base + d] = (vals[t] - mu) * rstd * g[d] + bta[d];
    }
}

// ---------------- masked mean pool ----------------
__global__ void pool_kernel(const float* __restrict__ x,     // [B*S, D]
                            const float* __restrict__ mask,  // [B*S]
                            float* __restrict__ pooled)      // [B, D]
{
    int t = blockIdx.x * blockDim.x + threadIdx.x;
    if (t >= BATCH * D_MODEL) return;
    int d = t % D_MODEL;
    int b = t / D_MODEL;
    float acc = 0.0f;
    for (int s = 0; s < SEQ_LEN; ++s)
        acc += x[(size_t)(b * SEQ_LEN + s) * D_MODEL + d] * mask[b * SEQ_LEN + s];
    pooled[t] = acc * (1.0f / SEQ_LEN);
}

// ---------------- classifier: logits = pooled @ cls_W + cls_b ----------------
__global__ void cls_kernel(const float* __restrict__ pooled, // [B, D]
                           const float* __restrict__ Wc,     // [D, VOCAB]
                           const float* __restrict__ bc,     // [VOCAB]
                           float* __restrict__ logits)       // [B, VOCAB]
{
    int t = blockIdx.x * blockDim.x + threadIdx.x;
    if (t >= BATCH * VOCAB_SZ) return;
    int vcol = t % VOCAB_SZ;
    int b    = t / VOCAB_SZ;
    float acc = bc[vcol];
    const float* pr = pooled + (size_t)b * D_MODEL;
    #pragma unroll 4
    for (int kk = 0; kk < D_MODEL; ++kk)
        acc = fmaf(pr[kk], Wc[(size_t)kk * VOCAB_SZ + vcol], acc);
    logits[t] = acc;
}

// ---------------- host orchestration ----------------
extern "C" void kernel_launch(void* const* d_in, const int* in_sizes, int n_in,
                              void* d_out, int out_size, void* d_ws, size_t ws_size,
                              hipStream_t stream)
{
    (void)in_sizes; (void)n_in; (void)out_size; (void)ws_size;

    const int*   ids      = (const int*)  d_in[0];
    const float* amask    = (const float*)d_in[1];
    const float* tok_emb  = (const float*)d_in[2];
    const float* pos_emb  = (const float*)d_in[3];
    const float* Wq       = (const float*)d_in[4];
    const float* Wk       = (const float*)d_in[5];
    const float* Wv       = (const float*)d_in[6];
    const float* Wo       = (const float*)d_in[7];
    const float* bq       = (const float*)d_in[8];
    const float* bk       = (const float*)d_in[9];
    const float* bv       = (const float*)d_in[10];
    const float* bo       = (const float*)d_in[11];
    const float* Wu       = (const float*)d_in[12];
    const float* bu       = (const float*)d_in[13];
    const float* ln_g     = (const float*)d_in[14];
    const float* ln_b     = (const float*)d_in[15];
    const float* norm_g   = (const float*)d_in[16];
    const float* norm_b   = (const float*)d_in[17];
    const float* cls_W    = (const float*)d_in[18];
    const float* cls_b    = (const float*)d_in[19];
    float* logits         = (float*)d_out;

    // ---- carve workspace ----
    char* base = (char*)d_ws;
    size_t off = 0;
    auto carve = [&](size_t bytes) -> char* {
        char* p = base + off;
        off = (off + bytes + 255) & ~(size_t)255;
        return p;
    };
    const size_t MD_F32 = (size_t)M_ROWS * D_MODEL * sizeof(float);   // 8 MB
    const size_t MD_F16 = (size_t)M_ROWS * D_MODEL * sizeof(f16_t);   // 4 MB
    const size_t W_F16  = (size_t)D_MODEL * D_MODEL * sizeof(f16_t);  // 512 KB

    float* x     = (float*)carve(MD_F32);
    float* qb    = (float*)carve(MD_F32);
    float* kb    = (float*)carve(MD_F32);
    float* vb    = (float*)carve(MD_F32);
    float* aout  = (float*)carve(MD_F32);
    float* proj  = (float*)carve(MD_F32);
    f16_t* xh    = (f16_t*)carve(MD_F16);
    f16_t* ah    = (f16_t*)carve(MD_F16);
    f16_t* WqT   = (f16_t*)carve(W_F16);
    f16_t* WkT   = (f16_t*)carve(W_F16);
    f16_t* WvT   = (f16_t*)carve(W_F16);
    f16_t* WoT   = (f16_t*)carve(W_F16);
    float* ug    = (float*)carve((size_t)M_ROWS * N_HEADS * sizeof(float));
    float* pooled= (float*)carve((size_t)BATCH * D_MODEL * sizeof(float));

    const int TPB = 256;
    const int n_md    = M_ROWS * D_MODEL;
    const int blk_md  = (n_md + TPB - 1) / TPB;
    const int blk_w   = (D_MODEL * D_MODEL + TPB - 1) / TPB;
    const int gemm_waves = (M_ROWS / 32) * (D_MODEL / 32);       // 2048 waves
    const int blk_gemm   = gemm_waves / (TPB / 32);              // 256 blocks
    const int blk_attn   = (BATCH * N_HEADS * SEQ_LEN) / (TPB / 32);
    const int blk_ln     = M_ROWS / (TPB / 32);

    // 1. embeddings
    embed_kernel<<<blk_md, TPB, 0, stream>>>(ids, tok_emb, pos_emb, x);

    // 2. layers
    for (int l = 0; l < N_LAYERS; ++l) {
        const size_t wo  = (size_t)l * D_MODEL * D_MODEL;
        const size_t bo_ = (size_t)l * D_MODEL;

        cvt_f16_kernel<<<blk_md, TPB, 0, stream>>>(x, xh, n_md);
        wtrans_kernel<<<blk_w, TPB, 0, stream>>>(Wq + wo, WqT);
        wtrans_kernel<<<blk_w, TPB, 0, stream>>>(Wk + wo, WkT);
        wtrans_kernel<<<blk_w, TPB, 0, stream>>>(Wv + wo, WvT);
        wtrans_kernel<<<blk_w, TPB, 0, stream>>>(Wo + wo, WoT);

        gemm_wmma_kernel<<<blk_gemm, TPB, 0, stream>>>(xh, WqT, bq + bo_, qb,
                                                       M_ROWS, D_MODEL, D_MODEL);
        gemm_wmma_kernel<<<blk_gemm, TPB, 0, stream>>>(xh, WkT, bk + bo_, kb,
                                                       M_ROWS, D_MODEL, D_MODEL);
        gemm_wmma_kernel<<<blk_gemm, TPB, 0, stream>>>(xh, WvT, bv + bo_, vb,
                                                       M_ROWS, D_MODEL, D_MODEL);

        gate_kernel<<<(M_ROWS * N_HEADS + TPB - 1) / TPB, TPB, 0, stream>>>(
            x, Wu + (size_t)l * D_MODEL * N_HEADS, bu + (size_t)l * N_HEADS, ug);

        attn_kernel<<<blk_attn, TPB, 0, stream>>>(qb, kb, vb, ug, aout);

        cvt_f16_kernel<<<blk_md, TPB, 0, stream>>>(aout, ah, n_md);
        gemm_wmma_kernel<<<blk_gemm, TPB, 0, stream>>>(ah, WoT, bo + bo_, proj,
                                                       M_ROWS, D_MODEL, D_MODEL);

        ln_kernel<<<blk_ln, TPB, 0, stream>>>(x, proj, ln_g + bo_, ln_b + bo_, x);
    }

    // 3. final LN -> masked mean pool -> classifier
    ln_kernel<<<blk_ln, TPB, 0, stream>>>(x, nullptr, norm_g, norm_b, qb);
    pool_kernel<<<(BATCH * D_MODEL + TPB - 1) / TPB, TPB, 0, stream>>>(qb, amask, pooled);
    cls_kernel<<<(BATCH * VOCAB_SZ + TPB - 1) / TPB, TPB, 0, stream>>>(
        pooled, cls_W, cls_b, logits);
}
